// PolicyNet_35416300323635
// MI455X (gfx1250) — matrix operand order repaired and checked by reading, not compiled, not run on hardware
//
#include <hip/hip_runtime.h>
#include <hip/hip_bf16.h>

typedef __attribute__((ext_vector_type(16))) _Float16 v16h;
typedef __attribute__((ext_vector_type(8)))  _Float16 v8h;
typedef __attribute__((ext_vector_type(8)))  float    v8f;

#define GAT_ALPHA 0.2f

static __device__ __forceinline__ float lrelu(float x) { return x >= 0.f ? x : GAT_ALPHA * x; }
static __device__ __forceinline__ float eluf(float x)  { return x > 0.f ? x : (__expf(x) - 1.f); }

union V16H { v16h v; v8h h[2]; _Float16 e[16]; };

// ---------------------------------------------------------------------------
// Prep kernels
// ---------------------------------------------------------------------------

// Pack int32 adjacency [B,N,N] into bitmask words [B,N,N/32].
__global__ void k_pack_adj(const int* __restrict__ adj, unsigned* __restrict__ out, int total) {
    int idx = blockIdx.x * blockDim.x + threadIdx.x;
    if (idx >= total) return;
    const int* p = adj + (size_t)idx * 32;
    unsigned m = 0;
#pragma unroll
    for (int j = 0; j < 32; ++j) m |= (p[j] > 0 ? 1u : 0u) << j;
    out[idx] = m;
}

__global__ void k_f32_to_f16(const float* __restrict__ in, _Float16* __restrict__ out, int n) {
    int i = blockIdx.x * blockDim.x + threadIdx.x;
    if (i < n) out[i] = (_Float16)in[i];
}

// W_heads [8][128][64] f32 -> Wcat [128][512] f16 (head h occupies cols h*64..h*64+63)
__global__ void k_pack_wheads(const float* __restrict__ W, _Float16* __restrict__ out) {
    int i = blockIdx.x * blockDim.x + threadIdx.x;   // i = f*512 + col, total 65536
    int f = i >> 9, col = i & 511;
    int h = col >> 6, c = col & 63;
    out[i] = (_Float16)W[(h * 128 + f) * 64 + c];
}

// ---------------------------------------------------------------------------
// Generic WMMA GEMM: C[M,Nn] = A[M,K] * B[K,Nn], f16 in, f32 acc.
// One wave computes a 16x64 output tile (4 N-tiles). item = mtile*ngroups + ngroup.
// ---------------------------------------------------------------------------
template <int KBLOCKS, bool OUTF32>
__global__ void __launch_bounds__(256)
k_gemm16(const _Float16* __restrict__ A, const _Float16* __restrict__ B,
         void* __restrict__ C, int lda, int ldb, int ldc, int ngroups) {
    int wave = threadIdx.x >> 5, lane = threadIdx.x & 31;
    int item = blockIdx.x * 8 + wave;
    int mt = item / ngroups, ng = item % ngroups;
    int i0 = mt * 16, n0 = ng * 64;
    int m = lane & 15, hh = lane >> 4;

    const _Float16* arow = A + (size_t)(i0 + m) * lda;
    v8f c0 = {}, c1 = {}, c2 = {}, c3 = {};
#pragma unroll
    for (int kb = 0; kb < KBLOCKS; ++kb) {
        int k0 = kb * 32;
        V16H a;
        a.h[0] = *(const v8h*)(arow + k0 + hh * 8);        // e=0..7  -> K = hh*8 + e
        a.h[1] = *(const v8h*)(arow + k0 + 16 + hh * 8);   // e=8..15 -> K = 16 + hh*8 + (e-8)
        const _Float16* brow = B + (size_t)k0 * ldb + n0 + m;  // lane -> n = m, element s -> table-order k
        v16h b0, b1, b2, b3;
#pragma unroll
        for (int s = 0; s < 16; ++s) {
            int kk = (s < 8) ? (hh * 8 + s) : (16 + hh * 8 + (s - 8));
            const _Float16* p = brow + (size_t)kk * ldb;
            b0[s] = p[0]; b1[s] = p[16]; b2[s] = p[32]; b3[s] = p[48];
        }
        c0 = __builtin_amdgcn_wmma_f32_16x16x32_f16(false, a.v, false, b0, (short)0, c0, false, false);
        c1 = __builtin_amdgcn_wmma_f32_16x16x32_f16(false, a.v, false, b1, (short)0, c1, false, false);
        c2 = __builtin_amdgcn_wmma_f32_16x16x32_f16(false, a.v, false, b2, (short)0, c2, false, false);
        c3 = __builtin_amdgcn_wmma_f32_16x16x32_f16(false, a.v, false, b3, (short)0, c3, false, false);
    }
    // C/D layout: lane -> n = lane&15, VGPR v -> row = v + 8*(lane>>4)
    if (OUTF32) {
        float* out = (float*)C;
#pragma unroll
        for (int v = 0; v < 8; ++v) {
            size_t r = (size_t)(i0 + v + 8 * hh) * ldc;
            out[r + n0 + m]      = c0[v];
            out[r + n0 + 16 + m] = c1[v];
            out[r + n0 + 32 + m] = c2[v];
            out[r + n0 + 48 + m] = c3[v];
        }
    } else {
        _Float16* out = (_Float16*)C;
#pragma unroll
        for (int v = 0; v < 8; ++v) {
            size_t r = (size_t)(i0 + v + 8 * hh) * ldc;
            out[r + n0 + m]      = (_Float16)c0[v];
            out[r + n0 + 16 + m] = (_Float16)c1[v];
            out[r + n0 + 32 + m] = (_Float16)c2[v];
            out[r + n0 + 48 + m] = (_Float16)c3[v];
        }
    }
}

// ---------------------------------------------------------------------------
// Attention score dots: s1[h,b,n] = Wh[b,n,h*64:..]·a1_h ; s2 likewise
// ---------------------------------------------------------------------------
__global__ void k_sdots_heads(const _Float16* __restrict__ Wh, const float* __restrict__ a_heads,
                              float* __restrict__ s1, float* __restrict__ s2) {
    int idx = blockIdx.x * blockDim.x + threadIdx.x;   // (h*32+b)*512 + n, total 131072
    int n = idx & 511; int hb = idx >> 9; int b = hb & 31; int h = hb >> 5;
    const _Float16* row = Wh + ((size_t)(b * 512 + n)) * 512 + h * 64;
    const float* a1 = a_heads + h * 128;
    const float* a2 = a1 + 64;
    float x1 = 0.f, x2 = 0.f;
#pragma unroll
    for (int c = 0; c < 64; ++c) { float w = (float)row[c]; x1 += w * a1[c]; x2 += w * a2[c]; }
    s1[idx] = x1; s2[idx] = x2;
}

__global__ void k_sdots_out(const float* __restrict__ Wh2, const float* __restrict__ a_out,
                            float* __restrict__ s1, float* __restrict__ s2) {
    int idx = blockIdx.x * blockDim.x + threadIdx.x;   // b*512+n, total 16384
    const float* row = Wh2 + (size_t)idx * 128;
    float x1 = 0.f, x2 = 0.f;
#pragma unroll 16
    for (int c = 0; c < 128; ++c) { x1 += row[c] * a_out[c]; x2 += row[c] * a_out[128 + c]; }
    s1[idx] = x1; s2[idx] = x2;
}

// ---------------------------------------------------------------------------
// Layer-1 fused masked-softmax attention * Wh (per batch, per head), WMMA.
// One wave: 16 rows x all 512 cols of att, times Wh[b,:,h*64:h*64+64].
// grid = ((b*8+h)*4 + blkgrp), block = 256 (8 waves -> 8 row-tiles).
// Branchless P-tile generation; epilogue: scale by mask/rowsum, ELU, store f16.
// ---------------------------------------------------------------------------
__global__ void __launch_bounds__(256)
k_gat_att1(const unsigned* __restrict__ adjbits,
           const float* __restrict__ s1h, const float* __restrict__ s2h,
           const float* __restrict__ masks,
           const _Float16* __restrict__ Wh, _Float16* __restrict__ Xout) {
    int bid = blockIdx.x;
    int blkgrp = bid & 3; int bh = bid >> 2; int h = bh & 7; int b = bh >> 3;

    __shared__ float s2sh[512];
    const float* s2p = s2h + (size_t)(h * 32 + b) * 512;
    for (int t = threadIdx.x; t < 512; t += 256) s2sh[t] = s2p[t];
    __syncthreads();

    int wave = threadIdx.x >> 5, lane = threadIdx.x & 31;
    int rt = blkgrp * 8 + wave;
    int i0 = rt * 16;
    int m = lane & 15, hh = lane >> 4;
    int i = i0 + m;
    int n0 = h * 64;

    float s1v = s1h[(size_t)(h * 32 + b) * 512 + i];
    const unsigned* arow = adjbits + (size_t)(b * 512 + i) * 16;

    v8f c0 = {}, c1 = {}, c2 = {}, c3 = {};
    float rsum = 0.f;
#pragma unroll 4
    for (int kb = 0; kb < 16; ++kb) {
        unsigned word = arow[kb];
        // This lane's 16 s2 values: two aligned 8-float runs (ds_load_b128 x4)
        const float4* s2a = (const float4*)&s2sh[kb * 32 + hh * 8];
        const float4* s2b = (const float4*)&s2sh[kb * 32 + 16 + hh * 8];
        float4 sa0 = s2a[0], sa1 = s2a[1];
        float4 sb0 = s2b[0], sb1 = s2b[1];
        float sav[8] = { sa0.x, sa0.y, sa0.z, sa0.w, sa1.x, sa1.y, sa1.z, sa1.w };
        float sbv[8] = { sb0.x, sb0.y, sb0.z, sb0.w, sb1.x, sb1.y, sb1.z, sb1.w };

        V16H a;
        float rs = 0.f;
#pragma unroll
        for (int e = 0; e < 8; ++e) {
            float m0 = (float)((word >> (hh * 8 + e)) & 1u);         // bit for K = hh*8+e
            float m1 = (float)((word >> (16 + hh * 8 + e)) & 1u);    // bit for K = 16+hh*8+e
            float p0 = __expf(lrelu(s1v + sav[e])) * m0;             // branchless masked softmax num
            float p1 = __expf(lrelu(s1v + sbv[e])) * m1;
            rs += p0 + p1;
            a.e[e]     = (_Float16)p0;
            a.e[8 + e] = (_Float16)p1;
        }
        rsum += rs;

        const _Float16* brow = Wh + (size_t)(b * 512 + kb * 32) * 512 + n0 + m;
        v16h b0, b1, b2, b3;
#pragma unroll
        for (int s = 0; s < 16; ++s) {
            int kk = (s < 8) ? (hh * 8 + s) : (16 + hh * 8 + (s - 8));
            const _Float16* p = brow + (size_t)kk * 512;
            b0[s] = p[0]; b1[s] = p[16]; b2[s] = p[32]; b3[s] = p[48];
        }
        c0 = __builtin_amdgcn_wmma_f32_16x16x32_f16(false, a.v, false, b0, (short)0, c0, false, false);
        c1 = __builtin_amdgcn_wmma_f32_16x16x32_f16(false, a.v, false, b1, (short)0, c1, false, false);
        c2 = __builtin_amdgcn_wmma_f32_16x16x32_f16(false, a.v, false, b2, (short)0, c2, false, false);
        c3 = __builtin_amdgcn_wmma_f32_16x16x32_f16(false, a.v, false, b3, (short)0, c3, false, false);
    }
    // combine per-half-lane partial row sums (lanes L and L^16 hold the same row)
    rsum += __shfl_xor(rsum, 16, 32);
    float mask = masks[b * 512 + i];
    float scale = mask / fmaxf(rsum, 1e-30f);   // reference: elu(h' * mask); softmax denom
#pragma unroll
    for (int v = 0; v < 8; ++v) {
        int r = v + 8 * hh;
        float sc = __shfl(scale, r, 32);        // row-r scale (held at lane r)
        size_t ro = (size_t)(b * 512 + i0 + r) * 512;
        Xout[ro + n0 + m]      = (_Float16)eluf(c0[v] * sc);
        Xout[ro + n0 + 16 + m] = (_Float16)eluf(c1[v] * sc);
        Xout[ro + n0 + 32 + m] = (_Float16)eluf(c2[v] * sc);
        Xout[ro + n0 + 48 + m] = (_Float16)eluf(c3[v] * sc);
    }
}

// ---------------------------------------------------------------------------
// Output attention only at the selected row per batch (reference only reads
// x[b, node_order[b]]). One block (128 threads) per batch element.
// ---------------------------------------------------------------------------
__global__ void k_select_att(const unsigned* __restrict__ adjbits, const long long* __restrict__ node_order,
                             const float* __restrict__ s1o, const float* __restrict__ s2o,
                             const float* __restrict__ masks, const float* __restrict__ Wh2,
                             float* __restrict__ selected) {
    int b = blockIdx.x; int t = threadIdx.x;   // 128 threads = 128 output features
    __shared__ float w[512];
    int isel = (int)node_order[b];
    float s1v = s1o[b * 512 + isel];
    const unsigned* arow = adjbits + (size_t)(b * 512 + isel) * 16;
    for (int j = t; j < 512; j += 128) {
        float bitf = (float)((arow[j >> 5] >> (j & 31)) & 1u);
        w[j] = __expf(lrelu(s1v + s2o[b * 512 + j])) * bitf;
    }
    __syncthreads();
    float acc = 0.f, rs = 0.f;
    for (int j = 0; j < 512; ++j) {
        float wj = w[j];
        rs += wj;
        acc += wj * Wh2[(size_t)(b * 512 + j) * 128 + t];
    }
    float mask = masks[b * 512 + isel];
    selected[b * 128 + t] = (acc / fmaxf(rs, 1e-30f)) * mask * mask;  // mask applied twice in ref
}

// ---------------------------------------------------------------------------
// MLP head + softmax. One block (256 threads) per batch element.
// ---------------------------------------------------------------------------
__global__ void k_mlp(const float* __restrict__ selected, const float* __restrict__ work,
                      const float* __restrict__ subtask,
                      const float* __restrict__ fc1w, const float* __restrict__ fc1b,
                      const float* __restrict__ fc2w, const float* __restrict__ fc2b,
                      const float* __restrict__ fc3w, const float* __restrict__ fc3b,
                      float* __restrict__ out) {
    int b = blockIdx.x; int t = threadIdx.x;
    __shared__ float inf[192], h1[256], h2[256], o[512], red[256];
    if (t < 128)       inf[t] = selected[b * 128 + t];
    else if (t < 160)  inf[t] = work[b * 32 + (t - 128)];
    else if (t < 192)  inf[t] = subtask[b * 32 + (t - 160)];
    __syncthreads();
    { float a = fc1b[t]; for (int k = 0; k < 192; ++k) a += inf[k] * fc1w[k * 256 + t]; h1[t] = fmaxf(a, 0.f); }
    __syncthreads();
    { float a = fc2b[t]; for (int k = 0; k < 256; ++k) a += h1[k] * fc2w[k * 256 + t]; h2[t] = fmaxf(a, 0.f); }
    __syncthreads();
#pragma unroll
    for (int cc = 0; cc < 2; ++cc) {
        int col = t + cc * 256;
        float a = fc3b[col];
        for (int k = 0; k < 256; ++k) a += h2[k] * fc3w[k * 512 + col];
        o[col] = a;
    }
    __syncthreads();
    // softmax over 512
    red[t] = fmaxf(o[t], o[t + 256]); __syncthreads();
    for (int s = 128; s > 0; s >>= 1) { if (t < s) red[t] = fmaxf(red[t], red[t + s]); __syncthreads(); }
    float mx = red[0]; __syncthreads();
    float e0 = __expf(o[t] - mx), e1 = __expf(o[t + 256] - mx);
    red[t] = e0 + e1; __syncthreads();
    for (int s = 128; s > 0; s >>= 1) { if (t < s) red[t] += red[t + s]; __syncthreads(); }
    float inv = 1.f / red[0];
    out[b * 512 + t]       = e0 * inv;
    out[b * 512 + t + 256] = e1 * inv;
}

// ---------------------------------------------------------------------------
extern "C" void kernel_launch(void* const* d_in, const int* in_sizes, int n_in,
                              void* d_out, int out_size, void* d_ws, size_t ws_size,
                              hipStream_t stream) {
    const float*     graph_inf    = (const float*)d_in[0];
    const int*       graph_matrix = (const int*)d_in[1];
    const float*     masks        = (const float*)d_in[2];
    const long long* node_order   = (const long long*)d_in[3];
    const float*     work         = (const float*)d_in[4];
    const float*     subtask      = (const float*)d_in[5];
    const float*     W_heads      = (const float*)d_in[6];
    const float*     a_heads      = (const float*)d_in[7];
    const float*     W_out        = (const float*)d_in[8];
    const float*     a_out        = (const float*)d_in[9];
    const float*     fc1w = (const float*)d_in[10];
    const float*     fc1b = (const float*)d_in[11];
    const float*     fc2w = (const float*)d_in[12];
    const float*     fc2b = (const float*)d_in[13];
    const float*     fc3w = (const float*)d_in[14];
    const float*     fc3b = (const float*)d_in[15];

    char* ws = (char*)d_ws;
    size_t off = 0;
    unsigned*  adjbits = (unsigned*)(ws + off);  off += (size_t)32 * 512 * 16 * 4;   // 1 MB
    _Float16*  Xf16    = (_Float16*)(ws + off);  off += (size_t)16384 * 128 * 2;     // 4 MB
    _Float16*  Wcat    = (_Float16*)(ws + off);  off += (size_t)128 * 512 * 2;       // 128 KB
    _Float16*  Wout16  = (_Float16*)(ws + off);  off += (size_t)512 * 128 * 2;       // 128 KB
    _Float16*  Wh1     = (_Float16*)(ws + off);  off += (size_t)16384 * 512 * 2;     // 16 MB
    float*     s1h     = (float*)(ws + off);     off += (size_t)8 * 32 * 512 * 4;    // 512 KB
    float*     s2h     = (float*)(ws + off);     off += (size_t)8 * 32 * 512 * 4;    // 512 KB
    _Float16*  X1      = (_Float16*)(ws + off);  off += (size_t)16384 * 512 * 2;     // 16 MB
    float*     Wh2     = (float*)(ws + off);     off += (size_t)16384 * 128 * 4;     // 8 MB
    float*     s1o     = (float*)(ws + off);     off += (size_t)16384 * 4;
    float*     s2o     = (float*)(ws + off);     off += (size_t)16384 * 4;
    float*     selv    = (float*)(ws + off);     off += (size_t)32 * 128 * 4;

    // Prep: pack adjacency bits, convert activations/weights to f16
    k_pack_adj  <<<1024, 256, 0, stream>>>(graph_matrix, adjbits, 32 * 512 * 16);
    k_f32_to_f16<<<8192, 256, 0, stream>>>(graph_inf, Xf16, 16384 * 128);
    k_pack_wheads<<<256, 256, 0, stream>>>(W_heads, Wcat);
    k_f32_to_f16<<<256, 256, 0, stream>>>(W_out, Wout16, 512 * 128);

    // Layer-1 projection: Wh1[16384,512] = Xf16[16384,128] @ Wcat[128,512]
    k_gemm16<4, false><<<1024, 256, 0, stream>>>(Xf16, Wcat, (void*)Wh1, 128, 512, 512, 8);

    // per-head attention score scalars
    k_sdots_heads<<<512, 256, 0, stream>>>(Wh1, a_heads, s1h, s2h);

    // Layer-1 fused masked softmax-attention * Wh (8.6 GFLOP of WMMA)
    k_gat_att1<<<1024, 256, 0, stream>>>(adjbits, s1h, s2h, masks, Wh1, X1);

    // Output projection: Wh2[16384,128] = X1[16384,512] @ Wout16[512,128]
    k_gemm16<16, true><<<256, 256, 0, stream>>>(X1, Wout16, (void*)Wh2, 512, 128, 128, 2);

    // output-layer score scalars
    k_sdots_out<<<64, 256, 0, stream>>>(Wh2, a_out, s1o, s2o);

    // attention output only at selected node per batch
    k_select_att<<<32, 128, 0, stream>>>(adjbits, node_order, s1o, s2o, masks, Wh2, selv);

    // MLP + softmax -> d_out [32,512] f32
    k_mlp<<<32, 256, 0, stream>>>(selv, work, subtask, fc1w, fc1b, fc2w, fc2b, fc3w, fc3b, (float*)d_out);
}
